// LSTM_62173946577315
// MI455X (gfx1250) — matrix operand order repaired and checked by reading, not compiled, not run on hardware
//
#include <hip/hip_runtime.h>
#include <hip/hip_bf16.h>
#include <math.h>

// ---------------------------------------------------------------------------
// LSTM on MI455X (gfx1250): per-step fused-gate GEMM via v_wmma_f32_16x16x32_bf16
// z = [x_t, h_prev] @ W^T  (B=128, K=E+H=2048, N=4H=4096), then gate pointwise.
// Weights packed once into B-fragment-native bf16 layout (L2-resident, 16MB).
// Register-blocked 2x2 tiles per wave (4 WMMA / 8 b128 loads per k-tile) and
// split-K over the x/h segments (blockIdx.z) to double wave parallelism on
// the sequential critical path; partials summed in the pointwise kernel.
// ---------------------------------------------------------------------------

typedef __attribute__((ext_vector_type(16))) __bf16       v16bf;
typedef __attribute__((ext_vector_type(8)))  float        v8f;
typedef __attribute__((ext_vector_type(4)))  unsigned int v4u;

union Frag {
    v4u   u4[2];   // two 16-byte halves
    v16bf bf;      // 16 bf16 values (one WMMA A or B fragment per lane)
};

#define T_STEPS 256
#define B_N     128
#define E_N     1024
#define H_N     1024
#define K_N     2048      // E + H
#define N_N     4096      // 4H
#define KT_N    64        // K_N / 32  (k-tiles of the full GEMM)
#define KT_X    32        // k-tiles per segment (E/32 == H/32)

// float -> bf16, round-to-nearest-even
__device__ __forceinline__ unsigned short f2bf(float f) {
    unsigned int u = __float_as_uint(f);
    unsigned int r = u + 0x7FFFu + ((u >> 16) & 1u);
    return (unsigned short)(r >> 16);
}

__device__ __forceinline__ float sigmoidf_(float x) {
    return 1.0f / (1.0f + __expf(-x));
}

// ---------------------------------------------------------------------------
// Prep kernels
// ---------------------------------------------------------------------------

// Convert embeds (T*B*E fp32) -> bf16
__global__ void lstm_cvt_x(const float* __restrict__ x, unsigned short* __restrict__ xb, int n) {
    int i = blockIdx.x * blockDim.x + threadIdx.x;
    if (i < n) xb[i] = f2bf(x[i]);
}

// Pack W = concat(W_g, W_i, W_f, W_o) (each H x (E+H) fp32, row-major) into
// bf16 B-fragment layout: Wp[((n_tile*KT_N + k_tile)*32 + lane)*16 + e]
// lane = (khalf<<4)|ncol holds B[k][n] = W[n][k] for
//   n = n_tile*16 + ncol, k = k_tile*32 + khalf*16 + e
// -> each lane's fragment is 32 contiguous bytes (coalesced 1KB per wave).
__global__ void lstm_pack_w(const float* __restrict__ Wg, const float* __restrict__ Wi,
                            const float* __restrict__ Wf, const float* __restrict__ Wo,
                            unsigned short* __restrict__ Wp) {
    size_t id = (size_t)blockIdx.x * blockDim.x + threadIdx.x;   // N_N*K_N elements
    if (id >= (size_t)N_N * K_N) return;
    int    e     = (int)(id & 15);
    int    lane  = (int)((id >> 4) & 31);
    size_t tileI = id >> 9;                 // n_tile*KT_N + k_tile
    int    kt    = (int)(tileI & (KT_N - 1));
    int    ntile = (int)(tileI >> 6);
    int    n = ntile * 16 + (lane & 15);
    int    k = kt * 32 + ((lane >> 4) << 4) + e;
    int    gate = n >> 10;                  // n / H
    int    row  = n & (H_N - 1);
    const float* Wsrc = (gate == 0) ? Wg : (gate == 1) ? Wi : (gate == 2) ? Wf : Wo;
    Wp[id] = f2bf(Wsrc[(size_t)row * K_N + k]);
}

// Zero d_out[t=0], c state, and bf16 h state
__global__ void lstm_init(float* __restrict__ out0, float* __restrict__ c,
                          unsigned short* __restrict__ hb) {
    int i = blockIdx.x * blockDim.x + threadIdx.x;   // B_N*H_N
    out0[i] = 0.0f;
    c[i]    = 0.0f;
    hb[i]   = 0;
}

// ---------------------------------------------------------------------------
// Per-step GEMM (split-K): zp[seg] = A_seg (128x1024 bf16) @ W_seg, fp32 accum.
//   seg = blockIdx.z: 0 -> x_t columns [0,1024), 1 -> h columns [1024,2048).
// Block = 256 threads = 8 waves arranged 2(M) x 4(N); each wave computes a
// 2x2 block of 16x16 tiles (32x32 output patch). Block covers 64 rows x 128
// cols; grid = (32, 2, 2) -> 1024 waves per step.
// ---------------------------------------------------------------------------
__global__ void __launch_bounds__(256) lstm_gemm(
        const unsigned short* __restrict__ xt,   // B_N x E_N bf16 (this timestep)
        const unsigned short* __restrict__ h,    // B_N x H_N bf16 (prev hidden)
        const unsigned short* __restrict__ Wp,   // packed weights
        float* __restrict__ zp) {                // 2 x B_N x N_N fp32 partials
    const int lane = threadIdx.x & 31;
    const int wave = threadIdx.x >> 5;
    const int wn   = wave & 3;               // N wave group (0..3)
    const int wm   = wave >> 2;              // M wave group (0..1)
    const int seg  = blockIdx.z;             // 0: x segment, 1: h segment
    const int m_tile0 = blockIdx.y * 4 + wm * 2;   // this wave: m_tile0, m_tile0+1
    const int n_tile0 = blockIdx.x * 8 + wn * 2;   // this wave: n_tile0, n_tile0+1
    const int m    = lane & 15;              // A-matrix row within tile
    const int ksel = (lane >> 4) << 4;       // 0 or 16 (elements): A k-half select

    v8f acc00 = {}, acc01 = {}, acc10 = {}, acc11 = {};

    // Packed-weight fragment bases for the two n-tiles, at this segment's k-tiles
    const unsigned short* wp0 = Wp + (size_t)n_tile0 * KT_N * 512
                                   + (size_t)seg * KT_X * 512 + (size_t)lane * 16;
    const unsigned short* wp1 = wp0 + (size_t)KT_N * 512;

    // A rows for this segment (x_t and h are both B_N x 1024, row-major)
    const unsigned short* abase = (seg == 0) ? xt : h;
    const unsigned short* ar0 = abase + (size_t)(m_tile0 * 16 + m) * E_N + ksel;
    const unsigned short* ar1 = ar0 + 16 * E_N;   // second m-tile, +16 rows

    #pragma unroll 2
    for (int kt = 0; kt < KT_X; ++kt) {
        const int kbase = kt * 32;
        Frag a0, a1, b0, b1;
        a0.u4[0] = *(const v4u*)(ar0 + kbase);
        a0.u4[1] = *(const v4u*)(ar0 + kbase + 8);
        a1.u4[0] = *(const v4u*)(ar1 + kbase);
        a1.u4[1] = *(const v4u*)(ar1 + kbase + 8);
        const unsigned short* w0 = wp0 + (size_t)kt * 512;
        const unsigned short* w1 = wp1 + (size_t)kt * 512;
        b0.u4[0] = *(const v4u*)(w0);
        b0.u4[1] = *(const v4u*)(w0 + 8);
        b1.u4[0] = *(const v4u*)(w1);
        b1.u4[1] = *(const v4u*)(w1 + 8);
        acc00 = __builtin_amdgcn_wmma_f32_16x16x32_bf16(false, a0.bf, false, b0.bf, (short)0, acc00, false, false);
        acc01 = __builtin_amdgcn_wmma_f32_16x16x32_bf16(false, a0.bf, false, b1.bf, (short)0, acc01, false, false);
        acc10 = __builtin_amdgcn_wmma_f32_16x16x32_bf16(false, a1.bf, false, b0.bf, (short)0, acc10, false, false);
        acc11 = __builtin_amdgcn_wmma_f32_16x16x32_bf16(false, a1.bf, false, b1.bf, (short)0, acc11, false, false);
    }

    // C/D layout: lane<16 -> rows 0..7, lane>=16 -> rows 8..15; col = lane&15
    float* zs = zp + (size_t)seg * B_N * N_N;
    const int rsel = (lane >> 4) << 3;
    const int cof  = lane & 15;
    #pragma unroll
    for (int r = 0; r < 8; ++r) {
        const size_t row0 = (size_t)(m_tile0 * 16 + rsel + r) * N_N;
        const size_t row1 = (size_t)((m_tile0 + 1) * 16 + rsel + r) * N_N;
        zs[row0 + n_tile0 * 16 + cof]        = acc00[r];
        zs[row0 + (n_tile0 + 1) * 16 + cof]  = acc01[r];
        zs[row1 + n_tile0 * 16 + cof]        = acc10[r];
        zs[row1 + (n_tile0 + 1) * 16 + cof]  = acc11[r];
    }
}

// ---------------------------------------------------------------------------
// Per-step pointwise: sum split-K partials, gates + cell/hidden update.
// One thread per (b, h).
// ---------------------------------------------------------------------------
__global__ void lstm_pointwise(const float* __restrict__ zp,   // 2 x B_N x N_N
                               const float* __restrict__ bg, const float* __restrict__ bi,
                               const float* __restrict__ bff, const float* __restrict__ bo,
                               float* __restrict__ c, unsigned short* __restrict__ hb,
                               float* __restrict__ out_t) {
    const int id = blockIdx.x * blockDim.x + threadIdx.x;   // B_N*H_N
    const int b  = id >> 10;
    const int hh = id & (H_N - 1);
    const float* z0 = zp + (size_t)b * N_N;
    const float* z1 = z0 + (size_t)B_N * N_N;
    const float g = tanhf(z0[hh]           + z1[hh]           + bg[hh]);
    const float i = sigmoidf_(z0[H_N + hh]     + z1[H_N + hh]     + bi[hh]);
    const float f = sigmoidf_(z0[2 * H_N + hh] + z1[2 * H_N + hh] + bff[hh]);
    const float o = sigmoidf_(z0[3 * H_N + hh] + z1[3 * H_N + hh] + bo[hh]);
    const float cn = f * c[id] + i * g;
    c[id] = cn;
    const float hn = o * tanhf(cn);
    out_t[id] = hn;
    hb[id] = f2bf(hn);
}

// ---------------------------------------------------------------------------
// Host-side launch
// ---------------------------------------------------------------------------
extern "C" void kernel_launch(void* const* d_in, const int* in_sizes, int n_in,
                              void* d_out, int out_size, void* d_ws, size_t ws_size,
                              hipStream_t stream) {
    const float* embeds = (const float*)d_in[0];
    const float* Wg = (const float*)d_in[1];
    const float* Wi = (const float*)d_in[2];
    const float* Wf = (const float*)d_in[3];
    const float* Wo = (const float*)d_in[4];
    const float* bg = (const float*)d_in[5];
    const float* bi = (const float*)d_in[6];
    const float* bf = (const float*)d_in[7];
    const float* bo = (const float*)d_in[8];
    float* out = (float*)d_out;

    // Workspace layout (bytes):
    //   xb : T*B*E bf16     = 64 MiB
    //   Wp : 4H*K bf16      = 16 MiB
    //   hb : B*H bf16       = 256 KiB
    //   c  : B*H f32        = 512 KiB
    //   zp : 2 * B*4H f32   = 4 MiB   (split-K partials)
    char* ws = (char*)d_ws;
    unsigned short* xb = (unsigned short*)ws;                 ws += (size_t)T_STEPS * B_N * E_N * 2;
    unsigned short* Wp = (unsigned short*)ws;                 ws += (size_t)N_N * K_N * 2;
    unsigned short* hb = (unsigned short*)ws;                 ws += (size_t)B_N * H_N * 2;
    float*          c  = (float*)ws;                          ws += (size_t)B_N * H_N * 4;
    float*          zp = (float*)ws;

    // Prep: cast embeds, pack weights, zero state + out[0]
    {
        const int n = T_STEPS * B_N * E_N;
        lstm_cvt_x<<<(n + 255) / 256, 256, 0, stream>>>(embeds, xb, n);
    }
    {
        const size_t n = (size_t)N_N * K_N;
        lstm_pack_w<<<(unsigned)((n + 255) / 256), 256, 0, stream>>>(Wg, Wi, Wf, Wo, Wp);
    }
    lstm_init<<<(B_N * H_N) / 256, 256, 0, stream>>>(out, c, hb);

    // Sequential scan over timesteps (reference starts at t=1; out[0] = zeros)
    const dim3 ggrid(32, 2, 2);   // N: 32x128 cols; M: 2x64 rows; z: K segment
    for (int t = 1; t < T_STEPS; ++t) {
        const unsigned short* xt = xb + (size_t)t * B_N * E_N;
        lstm_gemm<<<ggrid, 256, 0, stream>>>(xt, hb, Wp, zp);
        lstm_pointwise<<<(B_N * H_N) / 256, 256, 0, stream>>>(
            zp, bg, bi, bf, bo, c, hb, out + (size_t)t * B_N * H_N);
    }
}
